// SCAAttention_58213986730281
// MI455X (gfx1250) — matrix operand order repaired and checked by reading, not compiled, not run on hardware
//
#include <hip/hip_runtime.h>
#include <hip/hip_bf16.h>
#include <stdint.h>

// SCAAttention for MI455X (gfx1250): bf16 WMMA GEMMs + fp32 softmax/norm.
// B=32 S=512 H=1024 NH=16 D=64.
#define BB  32
#define SS  512
#define HH  1024
#define NHH 16
#define DD  64
#define MM  (BB * SS)   // 16384 tokens

typedef __bf16 bf16;
typedef __attribute__((ext_vector_type(16))) __bf16 v16bf;
typedef __attribute__((ext_vector_type(8)))  __bf16 v8bf;
typedef __attribute__((ext_vector_type(8)))  float  v8f;

// ---------------------------------------------------------------------------
// WMMA fragment loaders (wave32, 16x16x32 bf16 shapes per CDNA5 ISA layouts).
// A (16Mx32K): lane L -> row M = L%16; K-half = L/16; regs 0..7 hold
//   K = half*8+0..7, regs 8..15 hold K = 16+half*8+0..7.
// B (32Kx16N): lane L -> col N = L%16; K-half = L/16; regs 0..7 hold
//   K = half*8+0..7, regs 8..15 hold K = 16+half*8+0..7 (same as A).
// ---------------------------------------------------------------------------
static __device__ __forceinline__ v16bf load_frag_a(const bf16* base, int ld, int m0, int k0) {
  int lane = threadIdx.x & 31;
  const bf16* p = base + (size_t)(m0 + (lane & 15)) * ld + k0 + ((lane >> 4) << 3);
  v8bf lo = *(const v8bf*)p;
  v8bf hi = *(const v8bf*)(p + 16);
  v16bf r;
#pragma unroll
  for (int i = 0; i < 8; ++i) { r[i] = lo[i]; r[i + 8] = hi[i]; }
  return r;
}

static __device__ __forceinline__ v16bf load_frag_b(const bf16* base, int ld, int n0, int k0) {
  int lane = threadIdx.x & 31;
  const bf16* p = base + (size_t)(n0 + (lane & 15)) * ld + k0 + ((lane >> 4) << 3);
  v8bf lo = *(const v8bf*)p;
  v8bf hi = *(const v8bf*)(p + 16);
  v16bf r;
#pragma unroll
  for (int i = 0; i < 8; ++i) { r[i] = lo[i]; r[i + 8] = hi[i]; }
  return r;
}

static __device__ __forceinline__ v8f wmma_bf16(v16bf a, v16bf b, v8f c) {
  // 8-arg form: (neg_a, A, neg_b, B, c_mod, C, reuse_a, reuse_b)
  return __builtin_amdgcn_wmma_f32_16x16x32_bf16(false, a, false, b, (short)0, c, false, false);
}

// CDNA5 async global->LDS DMA (ASYNCcnt tracked; 16B per lane).
static __device__ __forceinline__ void async_b128(unsigned ldsoff, const void* g) {
  asm volatile("global_load_async_to_lds_b128 %0, %1, off" :: "v"(ldsoff), "v"(g) : "memory");
}

// ---------------------------------------------------------------------------
// fp32 -> bf16 conversion
// ---------------------------------------------------------------------------
__global__ void __launch_bounds__(256) sca_f32_to_bf16(const float* __restrict__ in,
                                                       bf16* __restrict__ out, int n) {
  int i = blockIdx.x * 256 + threadIdx.x;
  if (i < n) out[i] = (bf16)in[i];
}

// ---------------------------------------------------------------------------
// QKV projection: out = hs @ W^T + b, then interleaved 2D-bbox RoPE.
// mode 0: Q (RoPE even/odd heads, * 1/sqrt(D));  1: K (RoPE);  2: V (plain).
// Output layout [B, NH, S, D] bf16.
//
// Block tile 128(M) x 64(N), K-step 32, double-buffered LDS staged with
// async-LDS DMA: 3 async b128/thread per stage, `s_wait_asynccnt 3` keeps one
// stage in flight behind the WMMAs. Each of the 8 waves computes a 32x32
// sub-tile (2x2 fragments -> 4 WMMAs per K-step) from LDS.
// ---------------------------------------------------------------------------
__global__ void __launch_bounds__(256) sca_qkv_gemm_rope(
    const bf16* __restrict__ A, const bf16* __restrict__ W, const float* __restrict__ bias,
    const int* __restrict__ bbox, const float* __restrict__ idxx, const float* __restrict__ idxy,
    bf16* __restrict__ out, int mode) {
  __shared__ __align__(16) bf16 Asm[2][128 * 32];  // [buf][row][k]
  __shared__ __align__(16) bf16 Bsm[2][64 * 32];   // [buf][col][k]

  int mB = blockIdx.x >> 4;   // MM/128 = 128
  int nB = blockIdx.x & 15;   // HH/64  = 16
  int tid = threadIdx.x;
  int w = tid >> 5, lane = tid & 31;
  int wm = w >> 1, wn = w & 1;

  // Per-thread staging slots: A row (2x16B), B col-quarter (1x16B).
  int arow = tid >> 1, akoff = (tid & 1) << 4;
  int bcol = tid >> 2, bkoff = (tid & 3) << 3;
  const bf16* Ag = A + (size_t)(mB * 128 + arow) * HH + akoff;
  const bf16* Bg = W + (size_t)(nB * 64 + bcol) * HH + bkoff;
  unsigned aoff[2], boff[2];
#pragma unroll
  for (int u = 0; u < 2; ++u) {
    aoff[u] = (unsigned)(uintptr_t)&Asm[u][arow * 32 + akoff];
    boff[u] = (unsigned)(uintptr_t)&Bsm[u][bcol * 32 + bkoff];
  }

  auto stage = [&](int buf, int kstep) {
    const bf16* ga = Ag + kstep * 32;
    async_b128(aoff[buf], (const void*)ga);
    async_b128(aoff[buf] + 16, (const void*)(ga + 8));
    async_b128(boff[buf], (const void*)(Bg + kstep * 32));
  };

  v8f acc[2][2] = {};
  auto compute = [&](int buf) {
    const bf16* Ab = &Asm[buf][0];
    const bf16* Bb = &Bsm[buf][0];
    v16bf a0 = load_frag_a(Ab, 32, wm * 32, 0);
    v16bf a1 = load_frag_a(Ab, 32, wm * 32 + 16, 0);
    v16bf b0 = load_frag_b(Bb, 32, wn * 32, 0);
    v16bf b1 = load_frag_b(Bb, 32, wn * 32 + 16, 0);
    acc[0][0] = wmma_bf16(a0, b0, acc[0][0]);
    acc[0][1] = wmma_bf16(a0, b1, acc[0][1]);
    acc[1][0] = wmma_bf16(a1, b0, acc[1][0]);
    acc[1][1] = wmma_bf16(a1, b1, acc[1][1]);
  };

  stage(0, 0);
  for (int i = 0; i < HH / 32 - 1; ++i) {
    stage((i + 1) & 1, i + 1);
    asm volatile("s_wait_asynccnt 3" ::: "memory");  // stage i landed (in-order)
    __syncthreads();
    compute(i & 1);
    __syncthreads();  // everyone done reading buf before it is re-staged
  }
  asm volatile("s_wait_asynccnt 0" ::: "memory");
  __syncthreads();
  compute((HH / 32 - 1) & 1);

  // -------- epilogue: bias + interleaved 2D RoPE, scatter to [B,NH,S,D] -----
  int m0 = mB * 128 + wm * 32;
  int n0 = nB * 64 + wn * 32;
  float scale = (mode == 0) ? 0.125f : 1.0f;   // 1/sqrt(64) folded into Q

#pragma unroll
  for (int j = 0; j < 2; ++j) {
    int n = n0 + j * 16 + (lane & 15);
    int h = n >> 6, d = n & 63;
    float bn = bias[n];
    int pair = d >> 1;            // RoPE pair index i in [0,32)
    int fj = pair >> 1;           // frequency index in [0,16)
    int use2 = (h & 1);           // odd heads use (x2,y2)
    int odd_pair = (pair & 1);    // i%2: 0 -> x*idx_x, 1 -> y*idx_y
    float fx = idxx[fj], fy = idxy[fj];
#pragma unroll
    for (int i = 0; i < 2; ++i) {
#pragma unroll
      for (int r = 0; r < 8; ++r) {
        int m = m0 + i * 16 + r + ((lane >> 4) << 3);
        int b_ = m >> 9, s_ = m & 511;
        float val = acc[i][j][r] + bn;
        float res;
        if (mode <= 1) {
          const int* bb = bbox + ((size_t)b_ * SS + s_) * 4;
          float cx = (float)bb[use2 ? 2 : 0];
          float cy = (float)bb[use2 ? 3 : 1];
          float ang = odd_pair ? cy * fy : cx * fx;
          float sn, cs;
          __sincosf(ang, &sn, &cs);
          float partner = __shfl_xor(val, 1, 32);  // d^1 lives in the adjacent lane
          res = (d & 1) ? (val * cs + partner * sn) : (val * cs - partner * sn);
        } else {
          res = val;
        }
        out[(((size_t)b_ * NHH + h) * SS + s_) * DD + d] = (bf16)(res * scale);
      }
    }
  }
}

// ---------------------------------------------------------------------------
// Flash attention: block = (b, h, 128 q-rows); 8 waves x 16 rows.
// K tile staged global->LDS with async-LDS DMA; V tile transposed through
// VGPRs (DMA can't reshape). Online softmax fp32; P re-laid out through LDS.
// ---------------------------------------------------------------------------
__global__ void __launch_bounds__(256) sca_flash_attn(
    const bf16* __restrict__ Q, const bf16* __restrict__ K, const bf16* __restrict__ V,
    const float* __restrict__ amask, bf16* __restrict__ ctx) {
  __shared__ __align__(16) bf16 Klds[32 * 64];     // [t][d]
  __shared__ __align__(16) bf16 Vt[64 * 32];       // [d][t]
  __shared__ __align__(16) bf16 Plds[8][16 * 32];  // per-wave [m][t]

  int bid = blockIdx.x;
  int qb = bid & 3;
  int h = (bid >> 2) & 15;
  int b = bid >> 6;
  int w = threadIdx.x >> 5;
  int lane = threadIdx.x & 31;
  int tid = threadIdx.x;

  const bf16* Qb = Q + ((size_t)(b * NHH + h) * SS) * DD;
  const bf16* Kb = K + ((size_t)(b * NHH + h) * SS) * DD;
  const bf16* Vb = V + ((size_t)(b * NHH + h) * SS) * DD;

  int qrow0 = qb * 128 + w * 16;
  v16bf qf0 = load_frag_a(Qb, DD, qrow0, 0);
  v16bf qf1 = load_frag_a(Qb, DD, qrow0, 32);

  v8f O[4];
  v8f zf = {};
#pragma unroll
  for (int t = 0; t < 4; ++t) O[t] = zf;
  float mrow[8], lrow[8];
#pragma unroll
  for (int r = 0; r < 8; ++r) { mrow[r] = -1e30f; lrow[r] = 0.f; }

  int trow = tid >> 3;
  int dc = (tid & 7) << 3;
  unsigned kldsoff = (unsigned)(uintptr_t)&Klds[trow * 64 + dc];  // LDS byte offset

  for (int t0 = 0; t0 < SS; t0 += 32) {
    __syncthreads();  // all waves done reading previous K/V tile
    {
      // K: async DMA straight into LDS (no VGPR round trip).
      async_b128(kldsoff, (const void*)(Kb + (size_t)(t0 + trow) * DD + dc));
      // V: load + transpose into [d][t] while the DMA flies.
      v8bf vv = *(const v8bf*)(Vb + (size_t)(t0 + trow) * DD + dc);
#pragma unroll
      for (int e = 0; e < 8; ++e) Vt[(dc + e) * 32 + trow] = vv[e];
      asm volatile("s_wait_asynccnt 0" ::: "memory");
    }
    __syncthreads();

    // scores: two 16x16 tiles over 32 keys, K-dim = 64 = 2 WMMAs each
    v8f s0 = zf, s1 = zf;
    {
      v16bf b0 = load_frag_b(Klds, 64, 0, 0);
      v16bf b1 = load_frag_b(Klds, 64, 0, 32);
      v16bf b2 = load_frag_b(Klds, 64, 16, 0);
      v16bf b3 = load_frag_b(Klds, 64, 16, 32);
      s0 = wmma_bf16(qf0, b0, s0);
      s0 = wmma_bf16(qf1, b1, s0);
      s1 = wmma_bf16(qf0, b2, s1);
      s1 = wmma_bf16(qf1, b3, s1);
    }

    float am0 = amask[b * SS + t0 + (lane & 15)];
    float am1 = amask[b * SS + t0 + 16 + (lane & 15)];
    float mk0 = (1.f - am0) * -10000.f;
    float mk1 = (1.f - am1) * -10000.f;

#pragma unroll
    for (int r = 0; r < 8; ++r) {
      float v0 = s0[r] + mk0;
      float v1 = s1[r] + mk1;
      float mx = fmaxf(v0, v1);
#pragma unroll
      for (int o = 1; o < 16; o <<= 1) mx = fmaxf(mx, __shfl_xor(mx, o, 16));
      float mnew = fmaxf(mrow[r], mx);
      float corr = __expf(mrow[r] - mnew);
      mrow[r] = mnew;
      v0 = __expf(v0 - mnew);
      v1 = __expf(v1 - mnew);
      float rs = v0 + v1;
#pragma unroll
      for (int o = 1; o < 16; o <<= 1) rs += __shfl_xor(rs, o, 16);
      lrow[r] = lrow[r] * corr + rs;
#pragma unroll
      for (int t = 0; t < 4; ++t) O[t][r] *= corr;
      int mrw = r + ((lane >> 4) << 3);
      Plds[w][mrw * 32 + (lane & 15)] = (bf16)v0;
      Plds[w][mrw * 32 + 16 + (lane & 15)] = (bf16)v1;
    }

    // O += P(16x32) @ V(32x64): 4 WMMAs (per-wave LDS, no barrier needed)
    v16bf pf = load_frag_a(&Plds[w][0], 32, 0, 0);
#pragma unroll
    for (int t = 0; t < 4; ++t) {
      v16bf bv = load_frag_b(Vt, 32, t * 16, 0);
      O[t] = wmma_bf16(pf, bv, O[t]);
    }
  }

  // ctx[b, s, h*64+d] = O / l  (bf16 for the output-projection GEMM)
#pragma unroll
  for (int r = 0; r < 8; ++r) {
    float inv = 1.f / lrow[r];
    int qr = qrow0 + r + ((lane >> 4) << 3);
    size_t g = (size_t)b * SS + qr;
#pragma unroll
    for (int t = 0; t < 4; ++t)
      ctx[g * HH + h * DD + t * 16 + (lane & 15)] = (bf16)(O[t][r] * inv);
  }
}

// ---------------------------------------------------------------------------
// Output projection + residual: y = ctx @ Wo^T + bo + hs  (fp32 into d_out)
// Same async double-buffered LDS pipeline as the QKV GEMM.
// ---------------------------------------------------------------------------
__global__ void __launch_bounds__(256) sca_out_proj(
    const bf16* __restrict__ Cx, const bf16* __restrict__ Wo, const float* __restrict__ bo,
    const float* __restrict__ hs, float* __restrict__ y) {
  __shared__ __align__(16) bf16 Asm[2][128 * 32];
  __shared__ __align__(16) bf16 Bsm[2][64 * 32];

  int mB = blockIdx.x >> 4;
  int nB = blockIdx.x & 15;
  int tid = threadIdx.x;
  int w = tid >> 5, lane = tid & 31;
  int wm = w >> 1, wn = w & 1;

  int arow = tid >> 1, akoff = (tid & 1) << 4;
  int bcol = tid >> 2, bkoff = (tid & 3) << 3;
  const bf16* Ag = Cx + (size_t)(mB * 128 + arow) * HH + akoff;
  const bf16* Bg = Wo + (size_t)(nB * 64 + bcol) * HH + bkoff;
  unsigned aoff[2], boff[2];
#pragma unroll
  for (int u = 0; u < 2; ++u) {
    aoff[u] = (unsigned)(uintptr_t)&Asm[u][arow * 32 + akoff];
    boff[u] = (unsigned)(uintptr_t)&Bsm[u][bcol * 32 + bkoff];
  }

  auto stage = [&](int buf, int kstep) {
    const bf16* ga = Ag + kstep * 32;
    async_b128(aoff[buf], (const void*)ga);
    async_b128(aoff[buf] + 16, (const void*)(ga + 8));
    async_b128(boff[buf], (const void*)(Bg + kstep * 32));
  };

  v8f acc[2][2] = {};
  auto compute = [&](int buf) {
    const bf16* Ab = &Asm[buf][0];
    const bf16* Bb = &Bsm[buf][0];
    v16bf a0 = load_frag_a(Ab, 32, wm * 32, 0);
    v16bf a1 = load_frag_a(Ab, 32, wm * 32 + 16, 0);
    v16bf b0 = load_frag_b(Bb, 32, wn * 32, 0);
    v16bf b1 = load_frag_b(Bb, 32, wn * 32 + 16, 0);
    acc[0][0] = wmma_bf16(a0, b0, acc[0][0]);
    acc[0][1] = wmma_bf16(a0, b1, acc[0][1]);
    acc[1][0] = wmma_bf16(a1, b0, acc[1][0]);
    acc[1][1] = wmma_bf16(a1, b1, acc[1][1]);
  };

  stage(0, 0);
  for (int i = 0; i < HH / 32 - 1; ++i) {
    stage((i + 1) & 1, i + 1);
    asm volatile("s_wait_asynccnt 3" ::: "memory");
    __syncthreads();
    compute(i & 1);
    __syncthreads();
  }
  asm volatile("s_wait_asynccnt 0" ::: "memory");
  __syncthreads();
  compute((HH / 32 - 1) & 1);

  int m0 = mB * 128 + wm * 32;
  int n0 = nB * 64 + wn * 32;
#pragma unroll
  for (int j = 0; j < 2; ++j) {
    int n = n0 + j * 16 + (lane & 15);
    float bn = bo[n];
#pragma unroll
    for (int i = 0; i < 2; ++i) {
#pragma unroll
      for (int r = 0; r < 8; ++r) {
        int m = m0 + i * 16 + r + ((lane >> 4) << 3);
        size_t idx = (size_t)m * HH + n;
        y[idx] = acc[i][j][r] + bn + hs[idx];
      }
    }
  }
}

// ---------------------------------------------------------------------------
// In-place RMSNorm over H=1024 per token row.
// ---------------------------------------------------------------------------
__global__ void __launch_bounds__(256) sca_rmsnorm(float* __restrict__ y,
                                                   const float* __restrict__ g) {
  __shared__ float red[8];
  __shared__ float stot;
  int row = blockIdx.x;
  int tid = threadIdx.x;
  float v[4];
  float ss = 0.f;
#pragma unroll
  for (int i = 0; i < 4; ++i) {
    v[i] = y[(size_t)row * HH + tid + i * 256];
    ss += v[i] * v[i];
  }
#pragma unroll
  for (int o = 1; o < 32; o <<= 1) ss += __shfl_xor(ss, o, 32);
  if ((tid & 31) == 0) red[tid >> 5] = ss;
  __syncthreads();
  if (tid == 0) {
    float t = 0.f;
#pragma unroll
    for (int i = 0; i < 8; ++i) t += red[i];
    stot = t;
  }
  __syncthreads();
  float inv = rsqrtf(stot * (1.0f / (float)HH) + 1e-5f);
#pragma unroll
  for (int i = 0; i < 4; ++i) {
    int col = tid + i * 256;
    y[(size_t)row * HH + col] = g[col] * v[i] * inv;
  }
}

// ---------------------------------------------------------------------------
extern "C" void kernel_launch(void* const* d_in, const int* in_sizes, int n_in,
                              void* d_out, int out_size, void* d_ws, size_t ws_size,
                              hipStream_t stream) {
  (void)in_sizes; (void)n_in; (void)out_size; (void)ws_size;
  const float* hs   = (const float*)d_in[0];
  const int*   bbox = (const int*)d_in[1];
  const float* am   = (const float*)d_in[2];
  const float* Wq   = (const float*)d_in[3];
  const float* bq   = (const float*)d_in[4];
  const float* Wk   = (const float*)d_in[5];
  const float* bk   = (const float*)d_in[6];
  const float* Wv   = (const float*)d_in[7];
  const float* bv   = (const float*)d_in[8];
  const float* Wo   = (const float*)d_in[9];
  const float* bo   = (const float*)d_in[10];
  const float* rms  = (const float*)d_in[11];
  const float* ix   = (const float*)d_in[12];
  const float* iy   = (const float*)d_in[13];
  float* out = (float*)d_out;

  // Workspace carve-out (~168 MB total).
  char* p = (char*)d_ws;
  bf16* hsb = (bf16*)p; p += (size_t)MM * HH * 2;
  bf16* wqb = (bf16*)p; p += (size_t)HH * HH * 2;
  bf16* wkb = (bf16*)p; p += (size_t)HH * HH * 2;
  bf16* wvb = (bf16*)p; p += (size_t)HH * HH * 2;
  bf16* wob = (bf16*)p; p += (size_t)HH * HH * 2;
  bf16* Qb  = (bf16*)p; p += (size_t)MM * HH * 2;
  bf16* Kb  = (bf16*)p; p += (size_t)MM * HH * 2;
  bf16* Vb  = (bf16*)p; p += (size_t)MM * HH * 2;
  bf16* Cx  = (bf16*)p; p += (size_t)MM * HH * 2;

  const int nHS = MM * HH;
  const int nW  = HH * HH;
  sca_f32_to_bf16<<<(nHS + 255) / 256, 256, 0, stream>>>(hs, hsb, nHS);
  sca_f32_to_bf16<<<(nW + 255) / 256, 256, 0, stream>>>(Wq, wqb, nW);
  sca_f32_to_bf16<<<(nW + 255) / 256, 256, 0, stream>>>(Wk, wkb, nW);
  sca_f32_to_bf16<<<(nW + 255) / 256, 256, 0, stream>>>(Wv, wvb, nW);
  sca_f32_to_bf16<<<(nW + 255) / 256, 256, 0, stream>>>(Wo, wob, nW);

  const int gemmBlocks = (MM / 128) * (HH / 64);  // 2048 blocks, 8 waves each
  sca_qkv_gemm_rope<<<gemmBlocks, 256, 0, stream>>>(hsb, wqb, bq, bbox, ix, iy, Qb, 0);
  sca_qkv_gemm_rope<<<gemmBlocks, 256, 0, stream>>>(hsb, wkb, bk, bbox, ix, iy, Kb, 1);
  sca_qkv_gemm_rope<<<gemmBlocks, 256, 0, stream>>>(hsb, wvb, bv, bbox, ix, iy, Vb, 2);

  sca_flash_attn<<<BB * NHH * (SS / 128), 256, 0, stream>>>(Qb, Kb, Vb, am, Cx);

  sca_out_proj<<<gemmBlocks, 256, 0, stream>>>(Cx, wob, bo, hs, out);
  sca_rmsnorm<<<MM, 256, 0, stream>>>(out, rms);
}
